// LocallyConnected2d_44513041056551
// MI455X (gfx1250) — compile-verified, matrix operand order checked by reading, not run on hardware
//
#include <hip/hip_runtime.h>

typedef __attribute__((ext_vector_type(16))) __bf16 v16bf;
typedef __attribute__((ext_vector_type(8)))  __bf16 v8bf;
typedef __attribute__((ext_vector_type(8)))  float  v8f;

#define BATCH 64
#define CIN   64
#define COUT  64
#define HH    56
#define WW    56
#define HW    (HH*WW)        // 3136 spatial positions
#define NIT   (CIN/2)        // 32 chunks: 2 channels x 16 padded k-slots each
#define LDSS  40             // bf16 elems/row (32 + 8 pad) -> 80B rows, 16B aligned
#define BUFE  (BATCH*LDSS)   // elements per LDS buffer (5120 B)

__global__ __launch_bounds__(128)
void lc2d_wmma_kernel(const float* __restrict__ x,
                      const float* __restrict__ wgt,
                      float* __restrict__ out)
{
    // Double-buffered A (patches) and B (weights) tiles, bf16, k padded 9->16/channel
    __shared__ __bf16 As[2*BUFE];
    __shared__ __bf16 Bs[2*BUFE];

    const int pos   = blockIdx.x;        // spatial position (h*56+w)
    const int h     = pos / WW;
    const int w     = pos - h * WW;
    const int t     = threadIdx.x;       // 0..127 (4 waves)
    const int lane  = t & 31;
    const int wv    = t >> 5;            // wave id -> batch rows wv*16..+15
    const int nlo   = lane & 15;
    const int khalf = lane >> 4;         // K-half select per ISA A/B fragment layout
    const int mrow  = (wv << 4) + nlo;

    // Staging role: each thread owns one (row, channel-of-chunk) pair
    const int srow = t >> 1;             // A: batch row / B: cout column
    const int c2   = t & 1;              // which of the chunk's 2 channels

    // Block-uniform patch geometry, hoisted: clamped offsets + zero masks.
    int   off9[9];
    float msk9[9];
    #pragma unroll
    for (int kh = 0; kh < 3; ++kh) {
        const int ih  = h + kh - 1;
        const int ihc = ih < 0 ? 0 : (ih > HH-1 ? HH-1 : ih);
        #pragma unroll
        for (int kw = 0; kw < 3; ++kw) {
            const int iw  = w + kw - 1;
            const int iwc = iw < 0 ? 0 : (iw > WW-1 ? WW-1 : iw);
            const bool inb = ((unsigned)ih < (unsigned)HH) && ((unsigned)iw < (unsigned)WW);
            off9[kh*3+kw] = ihc * WW + iwc;
            msk9[kh*3+kw] = inb ? 1.0f : 0.0f;
        }
    }

    v8f acc[4] = {};                     // 16x64 output slice = 4 C tiles
    float av[9], bv[9];                  // staged fp32 values (registers)

    // All indices fit in 32 bits (x: 51MB, wgt: 462MB) -> 32-bit voffsets so the
    // backend can use SADDR+voffset global addressing (no 64-bit VGPR adds).
    auto load_chunk = [&](int it) {
        const int ci    = (it << 1) + c2;
        const unsigned abase = (unsigned)((srow * CIN + ci) * HW);
        #pragma unroll
        for (int r = 0; r < 9; ++r)
            av[r] = x[abase + (unsigned)off9[r]];     // clamped, always legal
        const unsigned bbase = (unsigned)(((ci * COUT + srow) * HW + pos) * 9);
        #pragma unroll
        for (int r = 0; r < 9; ++r)                   // 9 contiguous, single-use
            bv[r] = __builtin_nontemporal_load(&wgt[bbase + (unsigned)r]);
    };

    // Convert fp32 -> bf16 (masking OOB taps) and store: b128 (k0..7) + b16 (k8).
    auto store_chunk = [&](int buf) {
        v8bf a0, b0;
        #pragma unroll
        for (int r = 0; r < 8; ++r) {
            a0[r] = (__bf16)(av[r] * msk9[r]);
            b0[r] = (__bf16)bv[r];
        }
        __bf16* ad = &As[buf * BUFE + srow * LDSS + (c2 << 4)];
        __bf16* bd = &Bs[buf * BUFE + srow * LDSS + (c2 << 4)];
        *(v8bf*)ad = a0;
        ad[8] = (__bf16)(av[8] * msk9[8]);
        *(v8bf*)bd = b0;
        bd[8] = (__bf16)bv[8];
    };

    auto compute = [&](int buf) {
        const __bf16* Ab = &As[buf * BUFE];
        const __bf16* Bb = &Bs[buf * BUFE];
        // Hoist all fragments first so the 10 ds_loads can clause and the WMMAs
        // can proceed under partial dscnt waits.
        const v8bf alo = *(const v8bf*)&Ab[mrow * LDSS + (khalf << 3)];
        const v8bf ahi = *(const v8bf*)&Ab[mrow * LDSS + 16 + (khalf << 3)];
        const v16bf a = __builtin_shufflevector(alo, ahi,
            0,1,2,3,4,5,6,7,8,9,10,11,12,13,14,15);
        v16bf bfrag[4];
        #pragma unroll
        for (int nt = 0; nt < 4; ++nt) {
            const int jc = (nt << 4) + nlo;
            const v8bf blo = *(const v8bf*)&Bb[jc * LDSS + (khalf << 3)];
            const v8bf bhi = *(const v8bf*)&Bb[jc * LDSS + 16 + (khalf << 3)];
            bfrag[nt] = __builtin_shufflevector(blo, bhi,
                0,1,2,3,4,5,6,7,8,9,10,11,12,13,14,15);
        }
        #pragma unroll
        for (int nt = 0; nt < 4; ++nt)
            acc[nt] = __builtin_amdgcn_wmma_f32_16x16x32_bf16(
                false, a, false, bfrag[nt], (short)0, acc[nt], false, false);
    };

    // Padded k-slots 8..15 zeroed once for both buffers (slot 8 is overwritten
    // with real data every chunk; 9..15 stay zero forever).
    {
        const v8bf z = {};
        *(v8bf*)&As[0*BUFE + srow * LDSS + (c2 << 4) + 8] = z;
        *(v8bf*)&As[1*BUFE + srow * LDSS + (c2 << 4) + 8] = z;
        *(v8bf*)&Bs[0*BUFE + srow * LDSS + (c2 << 4) + 8] = z;
        *(v8bf*)&Bs[1*BUFE + srow * LDSS + (c2 << 4) + 8] = z;
    }

    // Software pipeline: prologue fills buffer 0
    load_chunk(0);
    store_chunk(0);
    __syncthreads();

    for (int it = 0; it < NIT; ++it) {
        const int cur = it & 1;
        if (it + 1 < NIT) load_chunk(it + 1);   // global loads overlap WMMA below
        compute(cur);
        if (it + 1 < NIT) store_chunk(cur ^ 1); // waits loads; writes other buffer
        __syncthreads();                        // one barrier per iteration
    }

    // Writeback: C/D layout -> lane N = lane&15, VGPR r -> M = r + 8*khalf
    const int mb = (wv << 4) + (khalf << 3);
    #pragma unroll
    for (int nt = 0; nt < 4; ++nt) {
        const int jc = (nt << 4) + nlo;
        #pragma unroll
        for (int r = 0; r < 8; ++r) {
            out[(size_t)((mb + r) * COUT + jc) * HW + (size_t)pos] = acc[nt][r];
        }
    }
}

extern "C" void kernel_launch(void* const* d_in, const int* in_sizes, int n_in,
                              void* d_out, int out_size, void* d_ws, size_t ws_size,
                              hipStream_t stream) {
    const float* x   = (const float*)d_in[0];   // (64,64,56,56) fp32
    const float* wgt = (const float*)d_in[1];   // (64,64,56,56,3,3) fp32
    float* out       = (float*)d_out;           // (64,64,56,56) fp32
    dim3 grid(HW);      // one workgroup per spatial position
    dim3 block(128);    // 4 waves
    hipLaunchKernelGGL(lc2d_wmma_kernel, grid, block, 0, stream, x, wgt, out);
}
